// CausalSelfAttention_61332132987331
// MI455X (gfx1250) — matrix operand order repaired and checked by reading, not compiled, not run on hardware
//
#include <hip/hip_runtime.h>
#include <hip/hip_bf16.h>

typedef _Float16 v16h __attribute__((ext_vector_type(16)));
typedef _Float16 v8h  __attribute__((ext_vector_type(8)));
typedef float    v8f  __attribute__((ext_vector_type(8)));

#define B_    4
#define T_    2048
#define C_    1024
#define TREF_ 512
#define H_    16
#define D_    64
#define S_    (T_ + TREF_)   /* 2560 */

__device__ __forceinline__ v8f wmma_f16(v16h a, v16h b, v8f c) {
  // D = A(16x32 f16) * B(32x16 f16) + C(f32)
  return __builtin_amdgcn_wmma_f32_16x16x32_f16(false, a, false, b, (short)0, c, false, false);
}

// 16-bit A-fragment: lane m = lane&15 holds row m; kb=(lane<16)?0:16
// halves[0..15] = A[m][kb..kb+15]
__device__ __forceinline__ v16h make_afrag(const _Float16* row, int kb) {
  v8h lo = *(const v8h*)(row + kb);
  v8h hi = *(const v8h*)(row + kb + 8);
  v16h r;
#pragma unroll
  for (int i = 0; i < 8; ++i) { r[i] = lo[i]; r[i + 8] = hi[i]; }
  return r;
}

// gfx1250 async copy: 16 bytes per lane, global -> LDS (ASYNCcnt-tracked).
// LDS byte offset = low 32 bits of generic LDS address (ISA aperture rule).
__device__ __forceinline__ void async_cp16(const _Float16* g, const _Float16* l) {
  asm volatile("global_load_async_to_lds_b128 %0, %1, off"
               :: "v"((unsigned)(uintptr_t)l),
                  "v"((unsigned long long)(uintptr_t)g)
               : "memory");
}
__device__ __forceinline__ void wait_async0() {
  asm volatile("s_wait_asynccnt 0x0" ::: "memory");
}

// ---------- prep kernels ----------
__global__ void __launch_bounds__(256)
cvt_f16(const float* __restrict__ in, _Float16* __restrict__ out) {
  size_t i = (size_t)blockIdx.x * 1024 + threadIdx.x;
#pragma unroll
  for (int j = 0; j < 4; ++j) out[i + 256 * j] = (_Float16)in[i + 256 * j];
}

// WT[n][k] = (f16) W[k][n]   (1024 x 1024)
__global__ void __launch_bounds__(256)
transpose_w(const float* __restrict__ W, _Float16* __restrict__ WT) {
  __shared__ float tile[32][33];
  const int bx = blockIdx.x * 32, by = blockIdx.y * 32;
  const int tx = threadIdx.x & 31, ty = threadIdx.x >> 5;   // 32 x 8
#pragma unroll
  for (int i = 0; i < 4; ++i) {
    int r = ty + i * 8;
    tile[r][tx] = W[(size_t)(by + r) * 1024 + bx + tx];
  }
  __syncthreads();
#pragma unroll
  for (int i = 0; i < 4; ++i) {
    int r = ty + i * 8;
    WT[(size_t)(bx + r) * 1024 + by + tx] = (_Float16)tile[tx][r];
  }
}

// ---------- GEMM: Y = X(f16)[M,1024] * WT(f16)[N,K]^T + bias ----------
// 128x128 block tile, 8 waves (4x2), wave tile 32x64, K stepped by 32,
// double-buffered async LDS staging.
// STORE: 0 = f32 row-major, 1 = f16 [bh][s][d] at s_off, 2 = f16 [bh][d][S] at s_off
template <int STORE>
__global__ void __launch_bounds__(256)
gemm_wmma(const _Float16* __restrict__ X, const _Float16* __restrict__ WT,
          const float* __restrict__ bias, void* __restrict__ out,
          int rows_per_batch, int s_total, int s_off) {
  __shared__ _Float16 aT[2][128 * 32];   // [buf][row][k]
  __shared__ _Float16 bT[2][128 * 32];   // [buf][col][k]
  const int tid = threadIdx.x;
  const int w = tid >> 5, lane = tid & 31;
  const int wM = w >> 1, wN = w & 1;
  const int m0 = blockIdx.y * 128, n0 = blockIdx.x * 128;
  const int lm = lane & 15, hi = lane >> 4;
  const int kbA = hi ? 16 : 0, kbB = hi ? 16 : 0;
  const int sr = (tid >> 2), sg = (tid & 3) * 8;           // staging row/segment
  const int sr2 = sr + 64;

  v8f acc[2][4];
#pragma unroll
  for (int i = 0; i < 2; ++i)
#pragma unroll
    for (int j = 0; j < 4; ++j) acc[i][j] = v8f{};

  // prologue: stage k-tile 0 into buffer 0
  {
    async_cp16(X  + (size_t)(m0 + sr ) * 1024 + sg, &aT[0][sr  * 32 + sg]);
    async_cp16(X  + (size_t)(m0 + sr2) * 1024 + sg, &aT[0][sr2 * 32 + sg]);
    async_cp16(WT + (size_t)(n0 + sr ) * 1024 + sg, &bT[0][sr  * 32 + sg]);
    async_cp16(WT + (size_t)(n0 + sr2) * 1024 + sg, &bT[0][sr2 * 32 + sg]);
  }
  wait_async0();
  __syncthreads();

  int buf = 0;
  for (int it = 0; it < 32; ++it) {
    if (it + 1 < 32) {                         // overlap: stage next tile
      int kn = (it + 1) * 32;
      async_cp16(X  + (size_t)(m0 + sr ) * 1024 + kn + sg, &aT[buf ^ 1][sr  * 32 + sg]);
      async_cp16(X  + (size_t)(m0 + sr2) * 1024 + kn + sg, &aT[buf ^ 1][sr2 * 32 + sg]);
      async_cp16(WT + (size_t)(n0 + sr ) * 1024 + kn + sg, &bT[buf ^ 1][sr  * 32 + sg]);
      async_cp16(WT + (size_t)(n0 + sr2) * 1024 + kn + sg, &bT[buf ^ 1][sr2 * 32 + sg]);
    }

    v16h aF[2], bF[4];
#pragma unroll
    for (int i = 0; i < 2; ++i)
      aF[i] = make_afrag(&aT[buf][(wM * 32 + i * 16 + lm) * 32], kbA);
#pragma unroll
    for (int j = 0; j < 4; ++j)
      bF[j] = *(const v16h*)&bT[buf][(wN * 64 + j * 16 + lm) * 32 + kbB];
#pragma unroll
    for (int i = 0; i < 2; ++i)
#pragma unroll
      for (int j = 0; j < 4; ++j)
        acc[i][j] = wmma_f16(aF[i], bF[j], acc[i][j]);

    wait_async0();
    __syncthreads();
    buf ^= 1;
  }

  // per-block batch index (128-row tile never crosses a batch: 128 | rows_per_batch)
  int bidx = 0, tbase = 0;
  if constexpr (STORE != 0) {
    bidx = m0 / rows_per_batch;
    tbase = m0 - bidx * rows_per_batch;
  }

#pragma unroll
  for (int i = 0; i < 2; ++i)
#pragma unroll
    for (int j = 0; j < 4; ++j)
#pragma unroll
      for (int r = 0; r < 8; ++r) {
        int lrow = wM * 32 + i * 16 + r + hi * 8;   // lanes16-31 hold M=r+8
        int col = n0 + wN * 64 + j * 16 + lm;
        float val = acc[i][j][r] + bias[col];
        if constexpr (STORE == 0) {
          ((float*)out)[(size_t)(m0 + lrow) * 1024 + col] = val;
        } else {
          int t = tbase + lrow;
          int h = col >> 6, d = col & 63;
          if constexpr (STORE == 1)
            ((_Float16*)out)[(((size_t)bidx * H_ + h) * s_total + s_off + t) * D_ + d] =
                (_Float16)val;
          else  // transposed value layout [bh][d][S]
            ((_Float16*)out)[((size_t)(bidx * H_ + h) * D_ + d) * S_ + s_off + t] =
                (_Float16)val;
        }
      }
}

// ---------- flash attention: 8 waves x 16 query rows, key chunks of 32 ----------
__global__ void __launch_bounds__(256)
attn_wmma(const _Float16* __restrict__ q, const _Float16* __restrict__ k,
          const _Float16* __restrict__ vT, _Float16* __restrict__ y) {
  __shared__ _Float16 vt[2][64 * 32];     // double-buffered V^T chunk: vt[buf][d][key]
  __shared__ _Float16 pL[8 * 16 * 32];    // per-wave P tile [16][32]
  const int tid = threadIdx.x;
  const int w = tid >> 5, lane = tid & 31;
  const int lm = lane & 15, hi = lane >> 4;
  const int kbA = hi ? 16 : 0, kbB = hi ? 16 : 0;
  const int bh = blockIdx.x;
  const int b = bh >> 4, h = bh & 15;
  const int qt0 = blockIdx.y * 128;
  const int qbase = qt0 + w * 16;
  const int sd = tid >> 2, ssg = (tid & 3) * 8;   // V staging coords
  const _Float16* vRow = vT + ((size_t)bh * D_ + sd) * S_ + ssg;

  const _Float16* qrow = q + ((size_t)bh * T_ + qbase + lm) * D_;
  v16h aQ0 = make_afrag(qrow, kbA);
  v16h aQ1 = make_afrag(qrow + 32, kbA);

  v8f o0 = {}, o1 = {}, o2 = {}, o3 = {};
  float mrun[8], lrun[8];
#pragma unroll
  for (int r = 0; r < 8; ++r) { mrun[r] = -3.0e38f; lrun[r] = 0.f; }

  const float scale = 0.125f;
  const int nSelf = (qt0 + 128) >> 5;     // self chunks needed by block (uniform)
  const int nC = nSelf + (TREF_ >> 5);    // + ref chunks
  _Float16* pW = &pL[w * 512];

  // prologue: stage chunk 0 (s0 = 0) into buffer 0
  async_cp16(vRow, &vt[0][sd * 32 + ssg]);
  wait_async0();
  __syncthreads();

  int buf = 0;
  for (int c = 0; c < nC; ++c) {
    const int s0 = (c < nSelf) ? c * 32 : T_ + (c - nSelf) * 32;
    if (c + 1 < nC) {                      // overlap: stage next chunk's V^T
      int s0n = (c + 1 < nSelf) ? (c + 1) * 32 : T_ + (c + 1 - nSelf) * 32;
      async_cp16(vRow + s0n, &vt[buf ^ 1][sd * 32 + ssg]);
    }

    // per-wave skip: chunk fully above this wave's causal limit
    if (!(s0 < T_ && s0 > qbase + 15)) {
      const _Float16* kr0 = k + ((size_t)bh * S_ + s0 + lm) * D_;
      const _Float16* kr1 = kr0 + 16 * D_;
      __builtin_prefetch(kr0 + 32 * D_, 0, 1);   // next chunk's K rows
      v8f c0 = {}, c1 = {};
      c0 = wmma_f16(aQ0, *(const v16h*)(kr0 + kbB),      c0);
      c0 = wmma_f16(aQ1, *(const v16h*)(kr0 + 32 + kbB), c0);
      c1 = wmma_f16(aQ0, *(const v16h*)(kr1 + kbB),      c1);
      c1 = wmma_f16(aQ1, *(const v16h*)(kr1 + 32 + kbB), c1);

      const bool needMask = (s0 < T_) && (s0 + 31 > qbase);
#pragma unroll
      for (int r = 0; r < 8; ++r) {
        int tq = qbase + r + hi * 8;
        float x0 = c0[r] * scale, x1 = c1[r] * scale;
        if (needMask) {                    // only boundary chunks pay this
          int sI0 = s0 + lm, sI1 = sI0 + 16;
          if (sI0 > tq) x0 = -3.0e38f;
          if (sI1 > tq) x1 = -3.0e38f;
        }
        float mx = fmaxf(x0, x1);
#pragma unroll
        for (int off = 1; off < 16; off <<= 1) mx = fmaxf(mx, __shfl_xor(mx, off, 32));
        float mnew = fmaxf(mrun[r], mx);
        float p0 = __expf(x0 - mnew), p1 = __expf(x1 - mnew);
        float rs = p0 + p1;
#pragma unroll
        for (int off = 1; off < 16; off <<= 1) rs += __shfl_xor(rs, off, 32);
        float alpha = __expf(mrun[r] - mnew);
        lrun[r] = lrun[r] * alpha + rs;
        mrun[r] = mnew;
        o0[r] *= alpha; o1[r] *= alpha; o2[r] *= alpha; o3[r] *= alpha;
        int prow = (r + hi * 8) * 32;
        pW[prow + lm]      = (_Float16)p0;
        pW[prow + 16 + lm] = (_Float16)p1;
      }
      asm volatile("s_wait_dscnt 0x0" ::: "memory");  // same-wave P RAW

      v16h aP = make_afrag(&pW[lm * 32], kbA);
      o0 = wmma_f16(aP, *(const v16h*)&vt[buf][(0 * 16 + lm) * 32 + kbB], o0);
      o1 = wmma_f16(aP, *(const v16h*)&vt[buf][(1 * 16 + lm) * 32 + kbB], o1);
      o2 = wmma_f16(aP, *(const v16h*)&vt[buf][(2 * 16 + lm) * 32 + kbB], o2);
      o3 = wmma_f16(aP, *(const v16h*)&vt[buf][(3 * 16 + lm) * 32 + kbB], o3);
    }

    wait_async0();
    __syncthreads();
    buf ^= 1;
  }

#pragma unroll
  for (int r = 0; r < 8; ++r) {
    int tq = qbase + r + hi * 8;
    float inv = 1.0f / lrun[r];
    size_t base = ((size_t)(b * T_ + tq)) * C_ + h * D_;
    y[base +  0 + lm] = (_Float16)(o0[r] * inv);
    y[base + 16 + lm] = (_Float16)(o1[r] * inv);
    y[base + 32 + lm] = (_Float16)(o2[r] * inv);
    y[base + 48 + lm] = (_Float16)(o3[r] * inv);
  }
}

extern "C" void kernel_launch(void* const* d_in, const int* in_sizes, int n_in,
                              void* d_out, int out_size, void* d_ws, size_t ws_size,
                              hipStream_t stream) {
  const float* x   = (const float*)d_in[0];
  const float* ref = (const float*)d_in[1];
  const float* Wq  = (const float*)d_in[2];  const float* bq  = (const float*)d_in[3];
  const float* Wk  = (const float*)d_in[4];  const float* bk  = (const float*)d_in[5];
  const float* Wv  = (const float*)d_in[6];  const float* bv  = (const float*)d_in[7];
  const float* Wrk = (const float*)d_in[8];  const float* brk = (const float*)d_in[9];
  const float* Wrv = (const float*)d_in[10]; const float* brv = (const float*)d_in[11];
  const float* Wp  = (const float*)d_in[12]; const float* bp  = (const float*)d_in[13];
  float* out = (float*)d_out;

  // f16 workspace layout
  _Float16* xh   = (_Float16*)d_ws;
  _Float16* refh = xh   + (size_t)B_ * T_ * C_;
  _Float16* WqT  = refh + (size_t)B_ * TREF_ * C_;
  _Float16* WkT  = WqT  + (size_t)C_ * C_;
  _Float16* WvT  = WkT  + (size_t)C_ * C_;
  _Float16* WrkT = WvT  + (size_t)C_ * C_;
  _Float16* WrvT = WrkT + (size_t)C_ * C_;
  _Float16* WpT  = WrvT + (size_t)C_ * C_;
  _Float16* qb   = WpT  + (size_t)C_ * C_;
  _Float16* kb   = qb   + (size_t)B_ * H_ * T_ * D_;
  _Float16* vb   = kb   + (size_t)B_ * H_ * S_ * D_;   // transposed [bh][d][S]
  _Float16* yb   = vb   + (size_t)B_ * H_ * S_ * D_;

  dim3 blk(256);

  // prep: fp32 -> f16 (x, ref), transpose+convert weights
  cvt_f16<<<dim3((B_ * T_ * C_) / 1024), blk, 0, stream>>>(x, xh);
  cvt_f16<<<dim3((B_ * TREF_ * C_) / 1024), blk, 0, stream>>>(ref, refh);
  dim3 gT(32, 32);
  transpose_w<<<gT, blk, 0, stream>>>(Wq,  WqT);
  transpose_w<<<gT, blk, 0, stream>>>(Wk,  WkT);
  transpose_w<<<gT, blk, 0, stream>>>(Wv,  WvT);
  transpose_w<<<gT, blk, 0, stream>>>(Wrk, WrkT);
  transpose_w<<<gT, blk, 0, stream>>>(Wrv, WrvT);
  transpose_w<<<gT, blk, 0, stream>>>(Wp,  WpT);

  dim3 gSelf(C_ / 128, (B_ * T_) / 128);     // 8 x 64
  dim3 gRef (C_ / 128, (B_ * TREF_) / 128);  // 8 x 16

  gemm_wmma<1><<<gSelf, blk, 0, stream>>>(xh,   WqT,  bq,  qb, T_,    T_, 0);
  gemm_wmma<1><<<gSelf, blk, 0, stream>>>(xh,   WkT,  bk,  kb, T_,    S_, 0);
  gemm_wmma<2><<<gSelf, blk, 0, stream>>>(xh,   WvT,  bv,  vb, T_,    S_, 0);
  gemm_wmma<1><<<gRef,  blk, 0, stream>>>(refh, WrkT, brk, kb, TREF_, S_, T_);
  gemm_wmma<2><<<gRef,  blk, 0, stream>>>(refh, WrvT, brv, vb, TREF_, S_, T_);

  dim3 gA(B_ * H_, T_ / 128);
  attn_wmma<<<gA, blk, 0, stream>>>(qb, kb, vb, yb);

  gemm_wmma<0><<<gSelf, blk, 0, stream>>>(yb, WpT, bp, out, T_, 0, 0);
}
